// ArcFaceHead_85899345920058
// MI455X (gfx1250) — matrix operand order repaired and checked by reading, not compiled, not run on hardware
//
#include <hip/hip_runtime.h>
#include <hip/hip_bf16.h>

// ---------------------------------------------------------------------------
// ArcFace head for MI455X (gfx1250, wave32, WMMA).
// bf16x3 split-precision GEMM on v_wmma_f32_16x16x32_bf16, f32 accumulate.
// ---------------------------------------------------------------------------

typedef __attribute__((ext_vector_type(16))) __bf16 bf16x16;
typedef __attribute__((ext_vector_type(8)))  float  f32x8;

constexpr int BM = 128;           // block tile M
constexpr int BN = 128;           // block tile N
constexpr int BK = 32;            // K per WMMA (bf16)
constexpr int PAD = 8;            // bf16 pad -> 80B row stride (16B aligned, bank-spread)
constexpr int SK = BK + PAD;

constexpr float ARC_S   = 64.0f;
constexpr float COS_M   = 0.87758256189037271f;   // cos(0.5)
constexpr float SIN_M   = 0.47942553860420300f;   // sin(0.5)
constexpr float ARC_TH  = -0.87758256189037271f;  // cos(pi - 0.5)
constexpr float ARC_MM  = 0.23971276930210150f;   // sin(pi - 0.5) * 0.5

// ---------------------------------------------------------------------------
// Pass 1: per-row 1/sqrt(sum(x^2)+eps).  One wave32 per row.
// ---------------------------------------------------------------------------
__global__ void rnorm_rows_kernel(const float* __restrict__ x,
                                  float* __restrict__ r,
                                  int rows, int cols) {
    const int wave = (int)((blockIdx.x * blockDim.x + threadIdx.x) >> 5);
    const int lane = threadIdx.x & 31;
    if (wave >= rows) return;
    const float4* row = reinterpret_cast<const float4*>(x + (size_t)wave * cols);
    const int nq = cols >> 2;
    float s = 0.0f;
    for (int i = lane; i < nq; i += 32) {
        float4 v = row[i];
        s += v.x * v.x + v.y * v.y + v.z * v.z + v.w * v.w;
    }
#pragma unroll
    for (int off = 16; off > 0; off >>= 1) s += __shfl_xor(s, off, 32);
    if (lane == 0) r[wave] = rsqrtf(s + 1e-12f);
}

// ---------------------------------------------------------------------------
// WMMA fragment loaders from LDS (K-contiguous rows), per ISA 7.12.2 tables.
//  A (16x32, lane = M):   lanes 0-15 hold K {0..7,16..23}; lanes 16-31 hold
//                         K {8..15,24..31}  -> two 16B chunks per lane.
//  B (32x16, lane = N):   same K order as A: lanes 0-15 hold K {0..7,16..23};
//                         lanes 16-31 hold K {8..15,24..31} -> two 16B chunks.
// ---------------------------------------------------------------------------
__device__ __forceinline__ bf16x16 load_a_frag(const __bf16* rowp, int half) {
    union { bf16x16 v; uint4 q[2]; } u;
    const uint4* p = reinterpret_cast<const uint4*>(rowp);
    u.q[0] = p[half];        // K = 8*half    .. 8*half+7
    u.q[1] = p[2 + half];    // K = 16+8*half .. 16+8*half+7
    return u.v;
}
__device__ __forceinline__ bf16x16 load_b_frag(const __bf16* rowp, int half) {
    union { bf16x16 v; uint4 q[2]; } u;
    const uint4* p = reinterpret_cast<const uint4*>(rowp);
    u.q[0] = p[half];
    u.q[1] = p[2 + half];
    return u.v;
}

// ---------------------------------------------------------------------------
// Pass 2: GEMM + normalize + clip + scale epilogue.
// out[m][n] = S * clip( (feat[m,:] . weight[n,:]) * ra[m] * rb[n], -1, 1 )
// ---------------------------------------------------------------------------
__global__ __launch_bounds__(256) void arcface_gemm_kernel(
    const float* __restrict__ A,    // features [Bdim][K]
    const float* __restrict__ W,    // weight   [C][K]
    const float* __restrict__ ra,   // [Bdim]
    const float* __restrict__ rb,   // [C]
    float* __restrict__ out,        // [Bdim][C]
    int C, int K) {

    __shared__ __bf16 sAhi[BM][SK];
    __shared__ __bf16 sAlo[BM][SK];
    __shared__ __bf16 sBhi[BN][SK];
    __shared__ __bf16 sBlo[BN][SK];

    const int m0   = blockIdx.y * BM;
    const int n0   = blockIdx.x * BN;
    const int tid  = threadIdx.x;
    const int wave = tid >> 5;
    const int lane = tid & 31;
    const int wm   = wave >> 2;        // 0..1  -> 64-row strip
    const int wn   = wave & 3;         // 0..3  -> 32-col strip
    const int half = lane >> 4;
    const int l16  = lane & 15;

    f32x8 acc[4][2] = {};              // 4 M-frags x 2 N-frags, f32 16x16

    const int lr = tid >> 3;           // 0..31 : staging row group
    const int lq = tid & 7;            // 0..7  : float4 within BK=32

    for (int k0 = 0; k0 < K; k0 += BK) {
        // ---- stage A tile (128 x 32 f32 -> bf16 hi/lo) -------------------
#pragma unroll
        for (int i = 0; i < 4; i++) {
            const int r = lr + 32 * i;
            float4 v = *reinterpret_cast<const float4*>(
                A + (size_t)(m0 + r) * K + k0 + 4 * lq);
            union { __bf16 h[4]; uint2 u; } hv, lv;
            hv.h[0] = (__bf16)v.x; lv.h[0] = (__bf16)(v.x - (float)hv.h[0]);
            hv.h[1] = (__bf16)v.y; lv.h[1] = (__bf16)(v.y - (float)hv.h[1]);
            hv.h[2] = (__bf16)v.z; lv.h[2] = (__bf16)(v.z - (float)hv.h[2]);
            hv.h[3] = (__bf16)v.w; lv.h[3] = (__bf16)(v.w - (float)hv.h[3]);
            *reinterpret_cast<uint2*>(&sAhi[r][4 * lq]) = hv.u;
            *reinterpret_cast<uint2*>(&sAlo[r][4 * lq]) = lv.u;
        }
        // ---- stage B tile (128 x 32 f32 -> bf16 hi/lo, N-edge zero-fill) -
#pragma unroll
        for (int i = 0; i < 4; i++) {
            const int r = lr + 32 * i;
            const int n = n0 + r;
            float4 v = make_float4(0.f, 0.f, 0.f, 0.f);
            if (n < C)
                v = *reinterpret_cast<const float4*>(
                    W + (size_t)n * K + k0 + 4 * lq);
            union { __bf16 h[4]; uint2 u; } hv, lv;
            hv.h[0] = (__bf16)v.x; lv.h[0] = (__bf16)(v.x - (float)hv.h[0]);
            hv.h[1] = (__bf16)v.y; lv.h[1] = (__bf16)(v.y - (float)hv.h[1]);
            hv.h[2] = (__bf16)v.z; lv.h[2] = (__bf16)(v.z - (float)hv.h[2]);
            hv.h[3] = (__bf16)v.w; lv.h[3] = (__bf16)(v.w - (float)hv.h[3]);
            *reinterpret_cast<uint2*>(&sBhi[r][4 * lq]) = hv.u;
            *reinterpret_cast<uint2*>(&sBlo[r][4 * lq]) = lv.u;
        }
        __syncthreads();

        // ---- load fragments ---------------------------------------------
        bf16x16 afh[4], afl[4], bfh[2], bfl[2];
#pragma unroll
        for (int i = 0; i < 4; i++) {
            const int r = wm * 64 + 16 * i + l16;
            afh[i] = load_a_frag(&sAhi[r][0], half);
            afl[i] = load_a_frag(&sAlo[r][0], half);
        }
#pragma unroll
        for (int j = 0; j < 2; j++) {
            const int r = wn * 32 + 16 * j + l16;
            bfh[j] = load_b_frag(&sBhi[r][0], half);
            bfl[j] = load_b_frag(&sBlo[r][0], half);
        }

        // ---- bf16x3: Ahi*Bhi + Alo*Bhi + Ahi*Blo, f32 accumulate --------
#pragma unroll
        for (int i = 0; i < 4; i++) {
#pragma unroll
            for (int j = 0; j < 2; j++) {
                acc[i][j] = __builtin_amdgcn_wmma_f32_16x16x32_bf16(
                    false, afh[i], false, bfh[j], (short)0, acc[i][j], false, false);
                acc[i][j] = __builtin_amdgcn_wmma_f32_16x16x32_bf16(
                    false, afl[i], false, bfh[j], (short)0, acc[i][j], false, false);
                acc[i][j] = __builtin_amdgcn_wmma_f32_16x16x32_bf16(
                    false, afh[i], false, bfl[j], (short)0, acc[i][j], false, false);
            }
        }
        __syncthreads();
    }

    // ---- epilogue: normalize, clip, scale, store -------------------------
    // C/D layout: VGPR e -> M = e (lanes 0-15) or e+8 (lanes 16-31), N = l16.
#pragma unroll
    for (int i = 0; i < 4; i++) {
#pragma unroll
        for (int j = 0; j < 2; j++) {
            const int n = n0 + wn * 32 + 16 * j + l16;
            if (n < C) {
                const float rbn = rb[n];
#pragma unroll
                for (int e = 0; e < 8; e++) {
                    const int m = m0 + wm * 64 + 16 * i + 8 * half + e;
                    float v = acc[i][j][e] * ra[m] * rbn;
                    v = fminf(1.0f, fmaxf(-1.0f, v));
                    out[(size_t)m * C + n] = ARC_S * v;
                }
            }
        }
    }
}

// ---------------------------------------------------------------------------
// Pass 3: margin fixup of the target column per row.
// ---------------------------------------------------------------------------
__global__ void arcface_fixup_kernel(const int* __restrict__ labels,
                                     float* __restrict__ out,
                                     int Bdim, int C) {
    const int b = blockIdx.x * blockDim.x + threadIdx.x;
    if (b >= Bdim) return;
    const int c = labels[b];
    const size_t idx = (size_t)b * C + c;
    const float cosv = out[idx] * (1.0f / ARC_S);     // already clipped
    float adj;
    if (cosv > ARC_TH) {
        // cos(acos(x) + m) = x cos m - sqrt(1 - x^2) sin m
        adj = cosv * COS_M - sqrtf(fmaxf(0.0f, 1.0f - cosv * cosv)) * SIN_M;
    } else {
        adj = cosv - ARC_MM;
    }
    out[idx] = ARC_S * adj;
}

// ---------------------------------------------------------------------------
extern "C" void kernel_launch(void* const* d_in, const int* in_sizes, int n_in,
                              void* d_out, int out_size, void* d_ws, size_t ws_size,
                              hipStream_t stream) {
    const float* feats  = (const float*)d_in[0];   // [B, D]
    const int*   labels = (const int*)  d_in[1];   // [B]
    const float* weight = (const float*)d_in[2];   // [C, D]
    float*       out    = (float*)d_out;           // [B, C]

    const int Bdim = in_sizes[1];
    const int Ddim = in_sizes[0] / Bdim;
    const int Cdim = in_sizes[2] / Ddim;

    float* ra = (float*)d_ws;          // [Bdim]
    float* rb = ra + Bdim;             // [Cdim]

    // per-row inverse norms (one wave per row)
    {
        int blocks = (Bdim * 32 + 255) / 256;
        rnorm_rows_kernel<<<blocks, 256, 0, stream>>>(feats, ra, Bdim, Ddim);
    }
    {
        int blocks = (Cdim * 32 + 255) / 256;
        rnorm_rows_kernel<<<blocks, 256, 0, stream>>>(weight, rb, Cdim, Ddim);
    }

    dim3 grid((Cdim + BN - 1) / BN, (Bdim + BM - 1) / BM);
    arcface_gemm_kernel<<<grid, 256, 0, stream>>>(feats, weight, ra, rb, out,
                                                  Cdim, Ddim);

    arcface_fixup_kernel<<<(Bdim + 255) / 256, 256, 0, stream>>>(labels, out,
                                                                 Bdim, Cdim);
}
